// Attention_49323404427915
// MI455X (gfx1250) — matrix operand order repaired and checked by reading, not compiled, not run on hardware
//
#include <hip/hip_runtime.h>
#include <hip/hip_bf16.h>

#define BB 2
#define TT 2048
#define DD 2048
#define NH_ 16
#define NKV_ 4
#define HD_ 128
#define BT (BB * TT)
#define KV (NKV_ * HD_)

typedef __bf16 bf16;
typedef __attribute__((ext_vector_type(16))) __bf16 v16bf;
typedef __attribute__((ext_vector_type(8)))  __bf16 v8bf;
typedef __attribute__((ext_vector_type(4)))  __bf16 v4bf;
typedef __attribute__((ext_vector_type(8)))  float  v8f;

union AB16 { v16bf v; v8bf h[2]; };

__device__ __forceinline__ bf16 f2bf(float f) {
  unsigned u = __builtin_bit_cast(unsigned, f);
  u += 0x7FFFu + ((u >> 16) & 1u);                 // round-to-nearest-even
  unsigned short s = (unsigned short)(u >> 16);
  return __builtin_bit_cast(bf16, s);
}

// A-fragment (16x32 bf16, row-major source, stride in elements).
// ISA layout: lane l holds row M=l&15; VGPR0-3 = K[kh*8 .. +8), VGPR4-7 = K[16+kh*8 .. +8), kh = l>>4.
__device__ __forceinline__ v16bf load_a16(const bf16* base, int stride) {
  const int l = threadIdx.x & 31;
  const bf16* p = base + (size_t)(l & 15) * stride + (l >> 4) * 8;
  AB16 f;
  f.h[0] = *(const v8bf*)(p);
  f.h[1] = *(const v8bf*)(p + 16);
  return f.v;
}

// B-fragment (32x16 bf16) for A @ W^T: column n of B = row n of W (contiguous).
// ISA layout: lane l holds col N=l&15; VGPR0-3 = K[kh*8 .. +8), VGPR4-7 = K[16+kh*8 .. +8), kh = l>>4.
__device__ __forceinline__ v16bf load_b16(const bf16* base, int stride) {
  const int l = threadIdx.x & 31;
  const bf16* p = base + (size_t)(l & 15) * stride + (l >> 4) * 8;
  AB16 f;
  f.h[0] = *(const v8bf*)(p);
  f.h[1] = *(const v8bf*)(p + 16);
  return f.v;
}

__device__ __forceinline__ v8f wmma_bf16(v16bf a, v16bf b, v8f c) {
  return __builtin_amdgcn_wmma_f32_16x16x32_bf16(false, a, false, b, (short)0, c, false, false);
}

// ---------------- elementwise f32 -> bf16 (x4 vectorized) ----------------
__global__ __launch_bounds__(256) void cvt_bf16_k(const float* __restrict__ s,
                                                  bf16* __restrict__ d, int n4) {
  int i = blockIdx.x * 256 + threadIdx.x;
  if (i >= n4) return;
  float4 f = ((const float4*)s)[i];
  v4bf o;
  o[0] = f2bf(f.x); o[1] = f2bf(f.y); o[2] = f2bf(f.z); o[3] = f2bf(f.w);
  ((v4bf*)d)[i] = o;
}

// ---------------- C[M,N] f32 = A[M,K]bf16 @ W[N,K]bf16 ^T ----------------
// 256 threads = 8 waves arranged 4(m) x 2(n); block tile 128x128.
// Wave tile 32x64: 8 WMMA per 12 x b128 loads (~21 FLOP/byte from cache).
__global__ __launch_bounds__(256) void gemm_bt(const bf16* __restrict__ A,
                                               const bf16* __restrict__ W,
                                               float* __restrict__ C,
                                               int M, int N, int K) {
  const int lane = threadIdx.x & 31;
  const int wave = threadIdx.x >> 5;
  const int m0 = blockIdx.y * 128 + (wave & 3) * 32;
  const int n0 = blockIdx.x * 128 + (wave >> 2) * 64;
  v8f acc[2][4] = {};
  for (int k0 = 0; k0 < K; k0 += 32) {
    v16bf a[2], b[4];
#pragma unroll
    for (int i = 0; i < 2; ++i)
      a[i] = load_a16(A + (size_t)(m0 + i * 16) * K + k0, K);
#pragma unroll
    for (int j = 0; j < 4; ++j)
      b[j] = load_b16(W + (size_t)(n0 + j * 16) * K + k0, K);
#pragma unroll
    for (int i = 0; i < 2; ++i)
#pragma unroll
      for (int j = 0; j < 4; ++j)
        acc[i][j] = wmma_bf16(a[i], b[j], acc[i][j]);
  }
  const int mb = (lane >> 4) * 8, nn = lane & 15;
#pragma unroll
  for (int i = 0; i < 2; ++i)
#pragma unroll
    for (int j = 0; j < 4; ++j)
#pragma unroll
      for (int r = 0; r < 8; ++r)
        C[(size_t)(m0 + i * 16 + mb + r) * N + (n0 + j * 16 + nn)] = acc[i][j][r];
}

// ---------------- per-(token, head) RMSNorm + RoPE, f32 in -> bf16 out ----------------
__global__ __launch_bounds__(256) void norm_rope_k(const float* __restrict__ in,
                                                   bf16* __restrict__ out,
                                                   const float* __restrict__ nw,
                                                   int heads) {
  const int lane = threadIdx.x & 31;
  const int wid = blockIdx.x * 8 + (threadIdx.x >> 5);
  if (wid >= BT * heads) return;
  const int h = wid % heads;
  const int row = wid / heads;
  const int t = row % TT;
  const int rs = heads * HD_;
  const float* p = in + (size_t)row * rs + h * HD_;
  float e0 = p[lane], e1 = p[lane + 32], e2 = p[lane + 64], e3 = p[lane + 96];
  float ss = e0 * e0 + e1 * e1 + e2 * e2 + e3 * e3;
#pragma unroll
  for (int off = 16; off; off >>= 1) ss += __shfl_xor(ss, off, 32);
  const float rr = rsqrtf(ss * (1.0f / HD_) + 1e-6f);
  float n0 = e0 * rr * nw[lane],      n1 = e1 * rr * nw[lane + 32];
  float n2 = e2 * rr * nw[lane + 64], n3 = e3 * rr * nw[lane + 96];
  const float CLT = 0.14391156516f;  // ln(10000)/64
  float a0 = (float)t * __expf(-(float)lane * CLT);
  float a1 = (float)t * __expf(-(float)(lane + 32) * CLT);
  float s0, c0, s1, c1;
  __sincosf(a0, &s0, &c0);
  __sincosf(a1, &s1, &c1);
  bf16* q = out + (size_t)row * rs + h * HD_;
  q[lane]      = f2bf(n0 * c0 - n2 * s0);
  q[lane + 32] = f2bf(n1 * c1 - n3 * s1);
  q[lane + 64] = f2bf(n2 * c0 + n0 * s0);
  q[lane + 96] = f2bf(n3 * c1 + n1 * s1);
}

// ---------------- V (BT, KV) f32 -> V^T (B, NKV, HD, T) bf16 ----------------
__global__ __launch_bounds__(256) void vtrans_k(const float* __restrict__ vf,
                                                bf16* __restrict__ vT) {
  size_t i = (size_t)blockIdx.x * 256 + threadIdx.x;  // over BT*KV exactly
  int col = (int)(i % KV);
  size_t row = i / KV;
  int b = (int)(row / TT), t = (int)(row % TT);
  int hk = col >> 7, d = col & 127;
  vT[(((size_t)b * NKV_ + hk) * HD_ + d) * TT + t] = f2bf(vf[i]);
}

// ---------------- causal flash attention, bf16 in/out, f32 online softmax ----------------
__global__ __launch_bounds__(128) void flash_k(const bf16* __restrict__ qb,
                                               const bf16* __restrict__ kb,
                                               const bf16* __restrict__ vT,
                                               bf16* __restrict__ attb) {
  __shared__ bf16 ldsP[4][16 * 32];  // per-wave P bounce tile
  const int lane = threadIdx.x & 31;
  const int wave = threadIdx.x >> 5;
  const int h = blockIdx.y, b = blockIdx.z;
  const int hk = h >> 2;  // NH/NKV = 4
  const int q0 = blockIdx.x * 64 + wave * 16;
  const bf16* Q  = qb + ((size_t)b * TT) * DD + h * HD_;
  const bf16* Kp = kb + ((size_t)b * TT) * KV + hk * HD_;
  const bf16* Vp = vT + ((size_t)b * NKV_ + hk) * HD_ * (size_t)TT;
  const int nn = lane & 15, kh = lane >> 4, mb = kh * 8;

  v16bf aq[4];  // Q rows q0..q0+15, 4 k-chunks of 32
#pragma unroll
  for (int c = 0; c < 4; ++c) {
    const bf16* p = Q + (size_t)(q0 + nn) * DD + c * 32 + kh * 8;
    AB16 f; f.h[0] = *(const v8bf*)p; f.h[1] = *(const v8bf*)(p + 16);
    aq[c] = f.v;
  }

  float mi[8], li[8];
  v8f o[8] = {};
#pragma unroll
  for (int r = 0; r < 8; ++r) { mi[r] = -1e30f; li[r] = 0.f; }

  const float SCALE = 0.08838834764831845f;  // 1/sqrt(128)
  const int nkt = (q0 + 16 + 31) / 32;       // 32-key tiles needed for causal rows
  for (int kt = 0; kt < nkt; ++kt) {
    v8f s[2];
#pragma unroll
    for (int half = 0; half < 2; ++half) {
      const int kbse = kt * 32 + half * 16;
      v8f acc = {};
#pragma unroll
      for (int c = 0; c < 4; ++c) {
        const bf16* p = Kp + (size_t)(kbse + nn) * KV + c * 32 + kh * 8;
        AB16 f; f.h[0] = *(const v8bf*)p; f.h[1] = *(const v8bf*)(p + 16);
        acc = wmma_bf16(aq[c], f.v, acc);
      }
#pragma unroll
      for (int r = 0; r < 8; ++r) acc[r] *= SCALE;
      if (kbse + 15 > q0) {  // tile touches/crosses the diagonal
        const int key = kbse + nn;
#pragma unroll
        for (int r = 0; r < 8; ++r)
          if (key > q0 + mb + r) acc[r] = -1e30f;
      }
      s[half] = acc;
    }
    // online softmax: rows live across 16-lane groups in the C layout
    float tmax[8], tsum[8], alpha[8];
#pragma unroll
    for (int r = 0; r < 8; ++r) {
      tmax[r] = fmaxf(s[0][r], s[1][r]);
#pragma unroll
      for (int off = 1; off < 16; off <<= 1)
        tmax[r] = fmaxf(tmax[r], __shfl_xor(tmax[r], off, 32));
    }
#pragma unroll
    for (int r = 0; r < 8; ++r) {
      float nm = fmaxf(mi[r], tmax[r]);
      alpha[r] = __expf(mi[r] - nm);
      mi[r] = nm;
      float e0 = __expf(s[0][r] - nm), e1 = __expf(s[1][r] - nm);
      s[0][r] = e0; s[1][r] = e1;
      tsum[r] = e0 + e1;
    }
#pragma unroll
    for (int r = 0; r < 8; ++r) {
#pragma unroll
      for (int off = 1; off < 16; off <<= 1)
        tsum[r] += __shfl_xor(tsum[r], off, 32);
      li[r] = li[r] * alpha[r] + tsum[r];
    }
#pragma unroll
    for (int d = 0; d < 8; ++d)
#pragma unroll
      for (int r = 0; r < 8; ++r) o[d][r] *= alpha[r];

    // bounce P (C layout) -> LDS -> A-fragment layout
    bf16* lp = &ldsP[wave][0];
#pragma unroll
    for (int r = 0; r < 8; ++r) {
      lp[(mb + r) * 32 + nn]      = f2bf(s[0][r]);
      lp[(mb + r) * 32 + 16 + nn] = f2bf(s[1][r]);
    }
    asm volatile("s_wait_dscnt 0" ::: "memory");  // cross-lane DS store->load in-wave
    v16bf ap;
    {
      const bf16* p = lp + (size_t)nn * 32 + kh * 8;
      AB16 f; f.h[0] = *(const v8bf*)p; f.h[1] = *(const v8bf*)(p + 16);
      ap = f.v;
    }
    asm volatile("s_wait_dscnt 0" ::: "memory");  // loads drained before next-iter stores

    // O += P (16x32) @ V (32x128) via V^T (dims-major -> contiguous B frags)
#pragma unroll
    for (int d = 0; d < 8; ++d) {
      const bf16* p = Vp + (size_t)(d * 16 + nn) * TT + kt * 32 + kh * 8;
      AB16 f; f.h[0] = *(const v8bf*)p; f.h[1] = *(const v8bf*)(p + 16);
      o[d] = wmma_bf16(ap, f.v, o[d]);
    }
  }
  float inv[8];
#pragma unroll
  for (int r = 0; r < 8; ++r) inv[r] = 1.0f / li[r];
#pragma unroll
  for (int d = 0; d < 8; ++d)
#pragma unroll
    for (int r = 0; r < 8; ++r)
      attb[(size_t)(b * TT + q0 + mb + r) * DD + h * HD_ + d * 16 + nn] =
          f2bf(o[d][r] * inv[r]);
}

extern "C" void kernel_launch(void* const* d_in, const int* in_sizes, int n_in,
                              void* d_out, int out_size, void* d_ws, size_t ws_size,
                              hipStream_t stream) {
  (void)in_sizes; (void)n_in; (void)out_size; (void)ws_size;
  const float* x   = (const float*)d_in[0];
  const float* wq  = (const float*)d_in[1];
  const float* wk  = (const float*)d_in[2];
  const float* wv  = (const float*)d_in[3];
  const float* wo  = (const float*)d_in[4];
  const float* qnw = (const float*)d_in[5];
  const float* knw = (const float*)d_in[6];

  char* ws = (char*)d_ws;
  auto take = [&](size_t bytes) -> char* {
    char* p = ws;
    ws += (bytes + 255) & ~(size_t)255;
    return p;
  };
  bf16*  xb  = (bf16*)take((size_t)BT * DD * 2);
  bf16*  wqb = (bf16*)take((size_t)DD * DD * 2);
  bf16*  wkb = (bf16*)take((size_t)KV * DD * 2);
  bf16*  wvb = (bf16*)take((size_t)KV * DD * 2);
  bf16*  wob = (bf16*)take((size_t)DD * DD * 2);
  float* qf  = (float*)take((size_t)BT * DD * 4);
  float* kf  = (float*)take((size_t)BT * KV * 4);
  float* vf  = (float*)take((size_t)BT * KV * 4);
  bf16*  qbb = (bf16*)take((size_t)BT * DD * 2);
  bf16*  kbb = (bf16*)take((size_t)BT * KV * 2);
  bf16*  vTb = (bf16*)take((size_t)BB * NKV_ * HD_ * TT * 2);
  bf16*  atb = (bf16*)take((size_t)BT * DD * 2);

  cvt_bf16_k<<<(BT * DD / 4 + 255) / 256, 256, 0, stream>>>(x,  xb,  BT * DD / 4);
  cvt_bf16_k<<<(DD * DD / 4 + 255) / 256, 256, 0, stream>>>(wq, wqb, DD * DD / 4);
  cvt_bf16_k<<<(KV * DD / 4 + 255) / 256, 256, 0, stream>>>(wk, wkb, KV * DD / 4);
  cvt_bf16_k<<<(KV * DD / 4 + 255) / 256, 256, 0, stream>>>(wv, wvb, KV * DD / 4);
  cvt_bf16_k<<<(DD * DD / 4 + 255) / 256, 256, 0, stream>>>(wo, wob, DD * DD / 4);

  gemm_bt<<<dim3(DD / 128, BT / 128), 256, 0, stream>>>(xb, wqb, qf, BT, DD, DD);
  gemm_bt<<<dim3(KV / 128, BT / 128), 256, 0, stream>>>(xb, wkb, kf, BT, KV, DD);
  gemm_bt<<<dim3(KV / 128, BT / 128), 256, 0, stream>>>(xb, wvb, vf, BT, KV, DD);

  norm_rope_k<<<(BT * NH_ + 7) / 8, 256, 0, stream>>>(qf, qbb, qnw, NH_);
  norm_rope_k<<<(BT * NKV_ + 7) / 8, 256, 0, stream>>>(kf, kbb, knw, NKV_);
  vtrans_k<<<(BT * KV) / 256, 256, 0, stream>>>(vf, vTb);

  flash_k<<<dim3(TT / 64, NH_, BB), 128, 0, stream>>>(qbb, kbb, vTb, atb);

  gemm_bt<<<dim3(DD / 128, BT / 128), 256, 0, stream>>>(atb, wob, (float*)d_out, BT, DD, DD);
}
